// DisentangledDynamics_42631845380583
// MI455X (gfx1250) — matrix operand order repaired and checked by reading, not compiled, not run on hardware
//
#include <hip/hip_runtime.h>

#define NB 8
#define TT 64
#define LATD 64

typedef __attribute__((ext_vector_type(16))) __bf16 v16bf;
typedef __attribute__((ext_vector_type(8)))  float  v8f;

__device__ __forceinline__ float sigmoidf_(float x){ return 1.f/(1.f+__expf(-x)); }
__device__ __forceinline__ float eluf_(float x){ return x > 0.f ? x : (__expf(x)-1.f); }
__device__ __forceinline__ float tanhf_(float x){
    float xc = fminf(fmaxf(x, -15.f), 15.f);
    float e  = __expf(2.f * xc);
    return (e - 1.f) / (e + 1.f);
}

// ---------------------------------------------------------------------------
// Generic strided/batched GEMM via v_wmma_f32_16x16x32_bf16.
// CONTRACT: M % 16 == 0, N % 64 == 0 (4 N-tiles per block), K % 32 == 0.
// Block = 128 threads = 4 waves; each wave computes one 16x16 tile of C.
// AVEC: sAk == 1 (A rows K-contiguous) -> float4 loads.
// BVEC: sBk == 1 (B cols K-contiguous) -> float4 loads.
// BIASM: 0=none 1=per-row(m) 2=per-col(n); ACT: 0=none 1=elu 2=tanh
// ---------------------------------------------------------------------------
template<bool AVEC, bool BVEC, int BIASM, int ACT>
__global__ void wmma_gemm(const float* __restrict__ A, const float* __restrict__ B,
                          float* __restrict__ C, int K,
                          long long sAm, long long sAk, long long aBat,
                          long long sBk, long long sBn, long long bBat,
                          long long sCm, long long sCn, long long cBat,
                          const float* __restrict__ bias)
{
    const float* Ab = A + (long long)blockIdx.z * aBat;
    const float* Bb = B + (long long)blockIdx.z * bBat;
    float*       Cb = C + (long long)blockIdx.z * cBat;
    int wave = threadIdx.x >> 5;
    int lane = threadIdx.x & 31;
    int r16  = lane & 15;
    int hi   = lane >> 4;
    int ma   = blockIdx.y * 16 + r16;                 // A row this lane supplies
    int nb   = (blockIdx.x * 4 + wave) * 16 + r16;    // B col this lane supplies

    const float* Ap = Ab + (long long)ma * sAm;       // row base
    const float* Bp = Bb + (long long)nb * sBn;       // col base

    v8f acc = {};
    for (int k0 = 0; k0 < K; k0 += 32) {
        v16bf av, bv;
        if (AVEC) {
            // A bf16 16x32 layout: this lane needs K = {hi*8..+7, 16+hi*8..+7}
            const float4 a0 = *reinterpret_cast<const float4*>(Ap + k0 + hi * 8);
            const float4 a1 = *reinterpret_cast<const float4*>(Ap + k0 + hi * 8 + 4);
            const float4 a2 = *reinterpret_cast<const float4*>(Ap + k0 + hi * 8 + 16);
            const float4 a3 = *reinterpret_cast<const float4*>(Ap + k0 + hi * 8 + 20);
            av[0]=(__bf16)a0.x; av[1]=(__bf16)a0.y; av[2]=(__bf16)a0.z; av[3]=(__bf16)a0.w;
            av[4]=(__bf16)a1.x; av[5]=(__bf16)a1.y; av[6]=(__bf16)a1.z; av[7]=(__bf16)a1.w;
            av[8]=(__bf16)a2.x; av[9]=(__bf16)a2.y; av[10]=(__bf16)a2.z; av[11]=(__bf16)a2.w;
            av[12]=(__bf16)a3.x; av[13]=(__bf16)a3.y; av[14]=(__bf16)a3.z; av[15]=(__bf16)a3.w;
        } else {
            #pragma unroll
            for (int j = 0; j < 16; ++j) {
                int ka = k0 + hi * 8 + (j < 8 ? j : j + 8);
                av[j] = (__bf16)Ap[(long long)ka * sAk];
            }
        }
        if (BVEC) {
            // B bf16 32x16 layout: this lane needs K = {hi*8..+7, 16+hi*8..+7}
            const float4 b0 = *reinterpret_cast<const float4*>(Bp + k0 + hi * 8);
            const float4 b1 = *reinterpret_cast<const float4*>(Bp + k0 + hi * 8 + 4);
            const float4 b2 = *reinterpret_cast<const float4*>(Bp + k0 + hi * 8 + 16);
            const float4 b3 = *reinterpret_cast<const float4*>(Bp + k0 + hi * 8 + 20);
            bv[0]=(__bf16)b0.x; bv[1]=(__bf16)b0.y; bv[2]=(__bf16)b0.z; bv[3]=(__bf16)b0.w;
            bv[4]=(__bf16)b1.x; bv[5]=(__bf16)b1.y; bv[6]=(__bf16)b1.z; bv[7]=(__bf16)b1.w;
            bv[8]=(__bf16)b2.x; bv[9]=(__bf16)b2.y; bv[10]=(__bf16)b2.z; bv[11]=(__bf16)b2.w;
            bv[12]=(__bf16)b3.x; bv[13]=(__bf16)b3.y; bv[14]=(__bf16)b3.z; bv[15]=(__bf16)b3.w;
        } else {
            #pragma unroll
            for (int j = 0; j < 16; ++j) {
                int kb = k0 + hi * 8 + (j < 8 ? j : j + 8);
                bv[j] = (__bf16)Bp[(long long)kb * sBk];
            }
        }
        acc = __builtin_amdgcn_wmma_f32_16x16x32_bf16(false, av, false, bv,
                                                      (short)0, acc, false, false);
    }
    int n = nb;
    float biasN = 0.f;
    if (BIASM == 2) biasN = bias[n];
    #pragma unroll
    for (int r = 0; r < 8; ++r) {
        int m = blockIdx.y * 16 + hi * 8 + r;
        float v = acc[r];
        if (BIASM == 1) v += bias[m];
        else if (BIASM == 2) v += biasN;
        if (ACT == 1) v = eluf_(v);
        else if (ACT == 2) v = tanhf_(v);
        Cb[(long long)m * sCm + (long long)n * sCn] = v;
    }
}

// ---------------------------------------------------------------------------
// Edge-conditioned spline graph conv: one block per edge.
// ---------------------------------------------------------------------------
__global__ void spatial_conv(const float* __restrict__ x, const int* __restrict__ ei,
                             const float* __restrict__ ea, const float* __restrict__ Wb,
                             float* __restrict__ agg, float* __restrict__ deg,
                             int E, int Vv, int Ci, int Co, int Tt)
{
    extern __shared__ float We[];
    int e   = blockIdx.x;
    int tid = threadIdx.x;
    int src = ei[e];
    int dst = ei[E + e];
    float w8[8]; int fl8[8];
    {
        float p0 = ea[3*e+0]*2.f, p1 = ea[3*e+1]*2.f, p2 = ea[3*e+2]*2.f;
        float c0f = fminf(fmaxf(floorf(p0), 0.f), 1.f);
        float c1f = fminf(fmaxf(floorf(p1), 0.f), 1.f);
        float c2f = fminf(fmaxf(floorf(p2), 0.f), 1.f);
        int i0 = (int)c0f, i1 = (int)c1f, i2 = (int)c2f;
        float f0 = p0 - c0f, f1 = p1 - c1f, f2 = p2 - c2f;
        int q = 0;
        #pragma unroll
        for (int a = 0; a < 2; ++a)
            #pragma unroll
            for (int b = 0; b < 2; ++b)
                #pragma unroll
                for (int c = 0; c < 2; ++c) {
                    w8[q]  = (a ? f0 : 1.f - f0) * (b ? f1 : 1.f - f1) * (c ? f2 : 1.f - f2);
                    fl8[q] = (i0 + a) * 9 + (i1 + b) * 3 + (i2 + c);
                    ++q;
                }
    }
    int CiCo = Ci * Co;
    for (int idx = tid; idx < CiCo; idx += blockDim.x) {
        float s = 0.f;
        #pragma unroll
        for (int q = 0; q < 8; ++q) s += w8[q] * Wb[(long long)fl8[q] * CiCo + idx];
        We[idx] = s;
    }
    if (tid == 0) atomicAdd(&deg[dst], 1.f);
    __syncthreads();
    int total = NB * Co * Tt;
    for (int idx = tid; idx < total; idx += blockDim.x) {
        int t = idx % Tt;
        int o = (idx / Tt) % Co;
        int n = idx / (Tt * Co);
        const float* xp = x + (((long long)n * Vv + src) * Ci) * Tt + t;
        float s = 0.f;
        for (int ci = 0; ci < Ci; ++ci) s += xp[(long long)ci * Tt] * We[ci * Co + o];
        atomicAdd(&agg[(((long long)n * Vv + dst) * Co + o) * Tt + t], s);
    }
}

__global__ void norm_elu(const float* __restrict__ agg, const float* __restrict__ deg,
                         float* __restrict__ out, int Vv, int Co, int Tt, long long total)
{
    long long idx = (long long)blockIdx.x * blockDim.x + threadIdx.x;
    if (idx >= total) return;
    long long v = (idx / ((long long)Co * Tt)) % Vv;
    float d = fmaxf(deg[v], 1.f);
    out[idx] = eluf_(agg[idx] / d);
}

__global__ void zero_kernel(float* __restrict__ p, long long n)
{
    long long i = (long long)blockIdx.x * blockDim.x + threadIdx.x;
    if (i < n) p[i] = 0.f;
}

__global__ void gru_step(const float* __restrict__ gi, const float* __restrict__ gh,
                         const float* __restrict__ bih, const float* __restrict__ bhh,
                         float* __restrict__ h)
{
    int idx = blockIdx.x * blockDim.x + threadIdx.x;   // 1024*64
    if (idx >= 1024 * LATD) return;
    int row = idx >> 6, o = idx & 63;
    const float* gir = gi + row * 192;
    const float* ghr = gh + row * 192;
    float ir = gir[o]       + bih[o],       hr = ghr[o]       + bhh[o];
    float iz = gir[64 + o]  + bih[64 + o],  hz = ghr[64 + o]  + bhh[64 + o];
    float ig = gir[128 + o] + bih[128 + o], hg = ghr[128 + o] + bhh[128 + o];
    float r = sigmoidf_(ir + hr);
    float z = sigmoidf_(iz + hz);
    float g = tanhf_(ig + r * hg);
    h[idx] = (1.f - z) * g + z * h[idx];
}

__global__ void trans_a(const float* __restrict__ rlin, const float* __restrict__ ulin,
                        const float* __restrict__ dR, const float* __restrict__ dU,
                        const float* __restrict__ br, const float* __restrict__ bu,
                        const float* __restrict__ zp, float* __restrict__ u,
                        float* __restrict__ rz)
{
    int idx = blockIdx.x * blockDim.x + threadIdx.x;
    if (idx >= 1024 * LATD) return;
    int o = idx & 63;
    float r = sigmoidf_(rlin[idx] + dR[idx] + br[o]);
    u[idx]  = sigmoidf_(ulin[idx] + dU[idx] + bu[o]);
    rz[idx] = r * zp[idx];
}

__global__ void trans_b(const float* __restrict__ clin, const float* __restrict__ dC,
                        const float* __restrict__ bc, const float* __restrict__ u,
                        float* __restrict__ zp, float* __restrict__ zs, int t)
{
    int idx = blockIdx.x * blockDim.x + threadIdx.x;
    if (idx >= 1024 * LATD) return;
    int o = idx & 63, row = idx >> 6;
    float c = tanhf_(clin[idx] + dC[idx] + bc[o]);
    float z = (1.f - u[idx]) * zp[idx] + u[idx] * c;
    zp[idx] = z;
    zs[((long long)row * LATD + o) * TT + t] = z;
}

// ---------------------------------------------------------------------------
extern "C" void kernel_launch(void* const* d_in, const int* in_sizes, int n_in,
                              void* d_out, int out_size, void* d_ws, size_t ws_size,
                              hipStream_t stream)
{
    (void)in_sizes; (void)n_in; (void)out_size; (void)ws_size;
    const int Vl[5] = {2048, 1024, 512, 256, 128};
    const int El[5] = {12288, 6144, 3072, 1536, 768};

    const float* x = (const float*)d_in[0];
    const int*   ei[5]; const float* ea[5];
    for (int l = 0; l < 5; ++l) { ei[l] = (const int*)d_in[1 + 2*l]; ea[l] = (const float*)d_in[2 + 2*l]; }
    const float* P01 = (const float*)d_in[11];
    const float* P12 = (const float*)d_in[12];
    const float* P23 = (const float*)d_in[13];
    const float* P34 = (const float*)d_in[14];
    const float* P43 = (const float*)d_in[15];
    const float* P32 = (const float*)d_in[16];
    const float* P21 = (const float*)d_in[17];
    const float* P10 = (const float*)d_in[18];
    const float* conv1_w   = (const float*)d_in[19];
    const float* conv2_w   = (const float*)d_in[20];
    const float* conv3_w   = (const float*)d_in[21];
    const float* conv4_w   = (const float*)d_in[22];
    const float* deconv4_w = (const float*)d_in[23];
    const float* deconv3_w = (const float*)d_in[24];
    const float* deconv2_w = (const float*)d_in[25];
    const float* deconv1_w = (const float*)d_in[26];
    const float* fce1_w = (const float*)d_in[27]; const float* fce1_b = (const float*)d_in[28];
    const float* fce2_w = (const float*)d_in[29]; const float* fce2_b = (const float*)d_in[30];
    const float* fcd3_w = (const float*)d_in[31]; const float* fcd3_b = (const float*)d_in[32];
    const float* fcd4_w = (const float*)d_in[33]; const float* fcd4_b = (const float*)d_in[34];
    const float* dom_Wih = (const float*)d_in[35]; const float* dom_Whh = (const float*)d_in[36];
    const float* dom_bih = (const float*)d_in[37]; const float* dom_bhh = (const float*)d_in[38];
    const float* dom_agg_w = (const float*)d_in[39]; const float* dom_agg_b = (const float*)d_in[40];
    const float* ini_Wih = (const float*)d_in[41]; const float* ini_Whh = (const float*)d_in[42];
    const float* ini_bih = (const float*)d_in[43]; const float* ini_bhh = (const float*)d_in[44];
    const float* ini_agg_w = (const float*)d_in[45]; const float* ini_agg_b = (const float*)d_in[46];
    const float* tr_Wr = (const float*)d_in[47]; const float* tr_br = (const float*)d_in[48];
    const float* tr_Wu = (const float*)d_in[49]; const float* tr_bu = (const float*)d_in[50];
    const float* tr_Wc = (const float*)d_in[51]; const float* tr_bc = (const float*)d_in[52];

    float* ws   = (float*)d_ws;
    float* bufA = ws;                       // 8,388,608 floats
    float* bufB = ws + 8388608LL;           // 8,388,608 floats
    float* sm   = ws + 16777216LL;
    float* giB   = sm; sm += 196608;
    float* ghB   = sm; sm += 196608;
    float* hB    = sm; sm += 65536;
    float* zDB   = sm; sm += 65536;
    float* zpB   = sm; sm += 65536;
    float* dRB   = sm; sm += 65536;
    float* dUB   = sm; sm += 65536;
    float* dCB   = sm; sm += 65536;
    float* rlinB = sm; sm += 65536;
    float* ulinB = sm; sm += 65536;
    float* clinB = sm; sm += 65536;
    float* uB    = sm; sm += 65536;
    float* rzB   = sm; sm += 65536;
    float* degB  = sm; sm += 2048;

    auto zero = [&](float* p, long long n) {
        zero_kernel<<<dim3((unsigned)((n + 255) / 256)), dim3(256), 0, stream>>>(p, n);
    };
    // N must be a multiple of 64, M of 16, K of 32 (true for all call sites).
    auto gemm = [&](const float* A, const float* B, float* C, int M, int N, int K,
                    long long sAm, long long sAk, long long aBat,
                    long long sBk, long long sBn, long long bBat,
                    long long sCm, long long sCn, long long cBat, int batch,
                    const float* bias, int bm, int act) {
        dim3 g(N / 64, M / 16, batch), b(128);
        bool avec = (sAk == 1), bvec = (sBk == 1);
#define GLAUNCH(AV, BV, BM, AC) \
        wmma_gemm<AV, BV, BM, AC><<<g, b, 0, stream>>>(A, B, C, K, sAm, sAk, aBat, \
            sBk, sBn, bBat, sCm, sCn, cBat, bias)
        if      ( avec && !bvec && bm == 0 && act == 0) GLAUNCH(true,  false, 0, 0);
        else if ( avec && !bvec && bm == 1 && act == 1) GLAUNCH(true,  false, 1, 1);
        else if ( avec && !bvec && bm == 1 && act == 2) GLAUNCH(true,  false, 1, 2);
        else if (!avec &&  bvec && bm == 0 && act == 0) GLAUNCH(false, true,  0, 0);
        else if ( avec &&  bvec && bm == 0 && act == 0) GLAUNCH(true,  true,  0, 0);
        else if ( avec &&  bvec && bm == 2 && act == 2) GLAUNCH(true,  true,  2, 2);
        else                                            GLAUNCH(false, false, 0, 0);
#undef GLAUNCH
    };
    auto spatial = [&](const float* xin, int lvl, const float* Wb, int Ci, int Co,
                       float* aggBuf, float* outBuf) {
        long long tot = 8LL * Vl[lvl] * Co * TT;
        zero(aggBuf, tot);
        zero(degB, Vl[lvl]);
        spatial_conv<<<dim3(El[lvl]), dim3(256), Ci * Co * sizeof(float), stream>>>(
            xin, ei[lvl], ea[lvl], Wb, aggBuf, degB, El[lvl], Vl[lvl], Ci, Co, TT);
        norm_elu<<<dim3((unsigned)((tot + 255) / 256)), dim3(256), 0, stream>>>(
            aggBuf, degB, outBuf, Vl[lvl], Co, TT, tot);
    };
    auto pool = [&](const float* P, const float* in, float* out, int Vo, int Vi, int CT) {
        gemm(P, in, out, Vo, CT, Vi,
             Vi, 1, 0,                       // A = P (shared)
             CT, 1, (long long)Vi * CT,      // B = h[n] (Vi x CT)
             CT, 1, (long long)Vo * CT, 8,   // C
             nullptr, 0, 0);
    };

    // ---------------- embedding ----------------
    spatial(x, 0, conv1_w, 1, 8, bufA, bufA);
    pool(P01, bufA, bufB, 1024, 2048, 8 * TT);
    spatial(bufB, 1, conv2_w, 8, 16, bufA, bufA);
    pool(P12, bufA, bufB, 512, 1024, 16 * TT);
    spatial(bufB, 2, conv3_w, 16, 32, bufA, bufA);
    pool(P23, bufA, bufB, 256, 512, 32 * TT);
    spatial(bufB, 3, conv4_w, 32, 64, bufA, bufA);
    pool(P34, bufA, bufB, 128, 256, 64 * TT);
    // fce1 (ELU) then fce2 (tanh) -> embed in bufB
    gemm(fce1_w, bufB, bufA, 128, TT, 64, 64, 1, 0, TT, 1, 64LL * TT,
         TT, 1, 128LL * TT, 1024, fce1_b, 1, 1);
    float* embedB = bufB;
    gemm(fce2_w, bufA, embedB, 64, TT, 128, 128, 1, 0, TT, 1, 128LL * TT,
         TT, 1, 64LL * TT, 1024, fce2_b, 1, 2);

    // ---------------- dom GRU (T steps) ----------------
    zero(hB, 1024LL * LATD);
    for (int t = 0; t < TT; ++t) {
        gemm(embedB + t, dom_Wih, giB, 1024, 192, 64, 64LL * TT, TT, 0, 1, 64, 0, 192, 1, 0, 1, nullptr, 0, 0);
        gemm(hB, dom_Whh, ghB, 1024, 192, 64, 64, 1, 0, 1, 64, 0, 192, 1, 0, 1, nullptr, 0, 0);
        gru_step<<<dim3(256), dim3(256), 0, stream>>>(giB, ghB, dom_bih, dom_bhh, hB);
    }
    gemm(hB, dom_agg_w, zDB, 1024, 64, 64, 64, 1, 0, 1, 64, 0, 64, 1, 0, 1, dom_agg_b, 2, 2);

    // ---------------- ini GRU (16 steps) ----------------
    zero(hB, 1024LL * LATD);
    for (int t = 0; t < 16; ++t) {
        gemm(embedB + t, ini_Wih, giB, 1024, 192, 64, 64LL * TT, TT, 0, 1, 64, 0, 192, 1, 0, 1, nullptr, 0, 0);
        gemm(hB, ini_Whh, ghB, 1024, 192, 64, 64, 1, 0, 1, 64, 0, 192, 1, 0, 1, nullptr, 0, 0);
        gru_step<<<dim3(256), dim3(256), 0, stream>>>(giB, ghB, ini_bih, ini_bhh, hB);
    }
    gemm(hB, ini_agg_w, zpB, 1024, 64, 64, 64, 1, 0, 1, 64, 0, 64, 1, 0, 1, ini_agg_b, 2, 2);

    // ---------------- latent transitions ----------------
    float* zsB = bufA;   // (1024, 64, T)
    gemm(zDB, tr_Wr + 64, dRB, 1024, 64, 64, 64, 1, 0, 1, 128, 0, 64, 1, 0, 1, nullptr, 0, 0);
    gemm(zDB, tr_Wu + 64, dUB, 1024, 64, 64, 64, 1, 0, 1, 128, 0, 64, 1, 0, 1, nullptr, 0, 0);
    gemm(zDB, tr_Wc + 64, dCB, 1024, 64, 64, 64, 1, 0, 1, 128, 0, 64, 1, 0, 1, nullptr, 0, 0);
    for (int t = 0; t < TT; ++t) {
        gemm(zpB, tr_Wr, rlinB, 1024, 64, 64, 64, 1, 0, 1, 128, 0, 64, 1, 0, 1, nullptr, 0, 0);
        gemm(zpB, tr_Wu, ulinB, 1024, 64, 64, 64, 1, 0, 1, 128, 0, 64, 1, 0, 1, nullptr, 0, 0);
        trans_a<<<dim3(256), dim3(256), 0, stream>>>(rlinB, ulinB, dRB, dUB, tr_br, tr_bu, zpB, uB, rzB);
        gemm(rzB, tr_Wc, clinB, 1024, 64, 64, 64, 1, 0, 1, 128, 0, 64, 1, 0, 1, nullptr, 0, 0);
        trans_b<<<dim3(256), dim3(256), 0, stream>>>(clinB, dCB, tr_bc, uB, zpB, zsB, t);
    }

    // ---------------- decoder ----------------
    gemm(fcd3_w, zsB, bufB, 128, TT, 64, 64, 1, 0, TT, 1, 64LL * TT,
         TT, 1, 128LL * TT, 1024, fcd3_b, 1, 1);
    gemm(fcd4_w, bufB, bufA, 64, TT, 128, 128, 1, 0, TT, 1, 128LL * TT,
         TT, 1, 64LL * TT, 1024, fcd4_b, 1, 1);
    pool(P43, bufA, bufB, 256, 128, 64 * TT);
    spatial(bufB, 3, deconv4_w, 64, 32, bufA, bufA);
    pool(P32, bufA, bufB, 512, 256, 32 * TT);
    spatial(bufB, 2, deconv3_w, 32, 16, bufA, bufA);
    pool(P21, bufA, bufB, 1024, 512, 16 * TT);
    spatial(bufB, 1, deconv2_w, 16, 8, bufA, bufA);
    pool(P10, bufA, bufB, 2048, 1024, 8 * TT);
    spatial(bufB, 0, deconv1_w, 8, 1, bufA, (float*)d_out);
}
